// changedGATConv_52037823758817
// MI455X (gfx1250) — matrix operand order repaired and checked by reading, not compiled, not run on hardware
//
#include <hip/hip_runtime.h>
#include <hip/hip_bf16.h>

#define N_NODES   50000
#define N_EDGES   1600000
#define IN_F      256
#define OUT_F     64
#define HEADS     4
#define EDGE_F    64
#define NUM_ET    8
#define NEG_SLOPE 0.2f
#define HD        (HEADS * OUT_F)   /* 256 */

typedef __attribute__((ext_vector_type(16))) __bf16 bf16x16;
typedef __attribute__((ext_vector_type(8)))  float  floatx8;

/* ---------------- workspace layout (in floats) ---------------- */
#define WS_FEATSRC 0
#define WS_EL      (N_NODES * HD)                      /* 12,800,000 */
#define WS_ER      (WS_EL  + N_NODES * HEADS)
#define WS_M       (WS_ER  + N_NODES * HEADS)
#define WS_DEN     (WS_M   + N_NODES * HEADS)
#define WS_TBL     (WS_DEN + N_NODES * HEADS)

/* ---- tiny: collapse edge_emb -> fc_e -> attn_e into an 8x4 table ---- */
__global__ void gat55_table_kernel(const float* __restrict__ fc_e_w,
                                   const float* __restrict__ edge_emb_w,
                                   const float* __restrict__ attn_e,
                                   float* __restrict__ tbl) {
    int tid = threadIdx.x;            /* 32 threads: (etype, head) pairs */
    if (tid >= NUM_ET * HEADS) return;
    int t = tid >> 2;                 /* edge type */
    int h = tid & 3;                  /* head */
    float acc = 0.0f;
    for (int f = 0; f < EDGE_F; ++f) {
        float proj = 0.0f;
        const float* wrow = fc_e_w + (h * EDGE_F + f) * EDGE_F;
        const float* erow = edge_emb_w + t * EDGE_F;
        for (int g = 0; g < EDGE_F; ++g) proj += wrow[g] * erow[g];
        acc += attn_e[h * EDGE_F + f] * proj;
    }
    tbl[t * HEADS + h] = acc;
}

/* ---- init: m=-inf, denom=0, rst=0 ---- */
__global__ void gat55_init_kernel(float* __restrict__ mmax,
                                  float* __restrict__ den,
                                  float* __restrict__ rst) {
    long long i = (long long)blockIdx.x * blockDim.x + threadIdx.x;
    if (i < (long long)N_NODES * HD) rst[i] = 0.0f;
    if (i < (long long)N_NODES * HEADS) {
        mmax[i] = -3.0e38f;
        den[i]  = 0.0f;
    }
}

/* ---- WMMA GEMM: feat_src[n, o] = sum_k feat[n,k] * fc_w[o,k] ----
 * One wave computes a 16(M) x 64(N) strip: 4 accumulators share the A tile.
 * K = 256 in 8 steps of 32 -> 32 v_wmma_f32_16x16x32_bf16 per wave.
 * 3125 M-tiles x 4 N-groups = 12500 waves. */
__global__ void gat55_gemm_wmma(const float* __restrict__ feat,
                                const float* __restrict__ fc_w,
                                float* __restrict__ feat_src) {
    const int lane = threadIdx.x & 31;
    const int gw   = blockIdx.x * 8 + (threadIdx.x >> 5);
    if (gw >= 3125 * 4) return;       /* uniform per-wave: EXEC stays all-ones */
    const int mtile = gw >> 2;
    const int ngrp  = gw & 3;
    const int lrow  = lane & 15;
    const int hi    = lane >> 4;
    const int row   = mtile * 16 + lrow;

    floatx8 acc0 = {0,0,0,0,0,0,0,0};
    floatx8 acc1 = acc0, acc2 = acc0, acc3 = acc0;
    floatx8 acc[4] = {acc0, acc1, acc2, acc3};

    #pragma unroll
    for (int k0 = 0; k0 < IN_F; k0 += 32) {
        /* A (16x32 bf16): lane holds row `row`; elems 0..7 -> K=k0+hi*8+j,
         * elems 8..15 -> K=k0+hi*8+16+j  (ISA 7.12.2 16-bit A layout) */
        const float* ap = feat + (long long)row * IN_F + k0 + hi * 8;
        bf16x16 a;
        #pragma unroll
        for (int j = 0; j < 8; ++j)  a[j]     = (__bf16)ap[j];
        #pragma unroll
        for (int j = 0; j < 8; ++j)  a[8 + j] = (__bf16)ap[16 + j];

        #pragma unroll
        for (int s = 0; s < 4; ++s) {
            /* B (32x16 bf16): lane holds column o; elems 0..7 -> K=k0+hi*8+j,
             * elems 8..15 -> K=k0+hi*8+16+j (same K order as A) */
            const int o = ngrp * 64 + s * 16 + lrow;
            const float* bp = fc_w + (long long)o * IN_F + k0 + hi * 8;
            bf16x16 b;
            #pragma unroll
            for (int j = 0; j < 8; ++j) { b[j] = (__bf16)bp[j]; b[8 + j] = (__bf16)bp[16 + j]; }
            acc[s] = __builtin_amdgcn_wmma_f32_16x16x32_bf16(
                false, a, false, b, (short)0, acc[s], false, false);
        }
    }

    /* C/D layout: VGPR v, lanes 0-15 -> M=v, lanes 16-31 -> M=v+8; N=lane&15 */
    #pragma unroll
    for (int s = 0; s < 4; ++s) {
        const int col = ngrp * 64 + s * 16 + lrow;
        #pragma unroll
        for (int v = 0; v < 8; ++v) {
            const int m = v + hi * 8;
            feat_src[(long long)(mtile * 16 + m) * HD + col] = acc[s][v];
        }
    }
}

/* ---- el / er per (node, head) ---- */
__global__ void gat55_elr_kernel(const float* __restrict__ feat_src,
                                 const float* __restrict__ attn_l,
                                 const float* __restrict__ attn_r,
                                 float* __restrict__ el,
                                 float* __restrict__ er) {
    int t = blockIdx.x * blockDim.x + threadIdx.x;
    if (t >= N_NODES * HEADS) return;
    int n = t >> 2, h = t & 3;
    const float* f = feat_src + (long long)n * HD + h * OUT_F;
    const float* al = attn_l + h * OUT_F;
    const float* ar = attn_r + h * OUT_F;
    float sl = 0.0f, sr = 0.0f;
    #pragma unroll 8
    for (int d = 0; d < OUT_F; ++d) { sl += f[d] * al[d]; sr += f[d] * ar[d]; }
    el[t] = sl;
    er[t] = sr;
}

__device__ __forceinline__ float gat55_lrelu(float x) {
    return x > 0.0f ? x : NEG_SLOPE * x;
}

/* ---- edge pass 1: logits + segment max via f32 atomics ---- */
__global__ void gat55_edge_max(const int* __restrict__ ef,
                               const int* __restrict__ src,
                               const int* __restrict__ dst,
                               const float* __restrict__ el,
                               const float* __restrict__ er,
                               const float* __restrict__ tbl,
                               float* __restrict__ mmax) {
    int t = blockIdx.x * blockDim.x + threadIdx.x;
    if (t >= N_EDGES) return;
    int s = src[t], d = dst[t], et = ef[t];
    float4 l4 = ((const float4*)el)[s];
    float4 r4 = ((const float4*)er)[d];
    float4 t4 = ((const float4*)tbl)[et];
    float v0 = gat55_lrelu(l4.x + r4.x + t4.x);
    float v1 = gat55_lrelu(l4.y + r4.y + t4.y);
    float v2 = gat55_lrelu(l4.z + r4.z + t4.z);
    float v3 = gat55_lrelu(l4.w + r4.w + t4.w);
    atomicMax(&mmax[d * 4 + 0], v0);
    atomicMax(&mmax[d * 4 + 1], v1);
    atomicMax(&mmax[d * 4 + 2], v2);
    atomicMax(&mmax[d * 4 + 3], v3);
}

/* ---- edge pass 2: ex = exp(e - m[dst]); denom += ex; stash ex in a_out ---- */
__global__ void gat55_edge_exp(const int* __restrict__ ef,
                               const int* __restrict__ src,
                               const int* __restrict__ dst,
                               const float* __restrict__ el,
                               const float* __restrict__ er,
                               const float* __restrict__ tbl,
                               const float* __restrict__ mmax,
                               float* __restrict__ den,
                               float* __restrict__ a_out) {
    int t = blockIdx.x * blockDim.x + threadIdx.x;
    if (t >= N_EDGES) return;
    int s = src[t], d = dst[t], et = ef[t];
    float4 l4 = ((const float4*)el)[s];
    float4 r4 = ((const float4*)er)[d];
    float4 t4 = ((const float4*)tbl)[et];
    float4 m4 = ((const float4*)mmax)[d];
    float4 ex;
    ex.x = __expf(gat55_lrelu(l4.x + r4.x + t4.x) - m4.x);
    ex.y = __expf(gat55_lrelu(l4.y + r4.y + t4.y) - m4.y);
    ex.z = __expf(gat55_lrelu(l4.z + r4.z + t4.z) - m4.z);
    ex.w = __expf(gat55_lrelu(l4.w + r4.w + t4.w) - m4.w);
    ((float4*)a_out)[t] = ex;
    atomicAdd(&den[d * 4 + 0], ex.x);
    atomicAdd(&den[d * 4 + 1], ex.y);
    atomicAdd(&den[d * 4 + 2], ex.z);
    atomicAdd(&den[d * 4 + 3], ex.w);
}

/* ---- edge pass 3: a = ex/denom[dst]; scatter a * feat_src[src] into rst ----
 * One wave per edge; each lane owns a contiguous 8-float slice of the 256-wide
 * message (two b128 loads, 8 f32 atomics). h = lane>>3 is slice-constant. */
__global__ void gat55_edge_agg(const int* __restrict__ src,
                               const int* __restrict__ dst,
                               const float* __restrict__ feat_src,
                               const float* __restrict__ den,
                               float* __restrict__ a_out,
                               float* __restrict__ rst) {
    const int lane = threadIdx.x & 31;
    const int edge = blockIdx.x * 8 + (threadIdx.x >> 5);
    if (edge >= N_EDGES) return;
    const int s = src[edge];
    const int d = dst[edge];
    float4 ex4 = ((const float4*)a_out)[edge];
    float4 dn4 = ((const float4*)den)[d];
    float av[4];
    av[0] = ex4.x / dn4.x;
    av[1] = ex4.y / dn4.y;
    av[2] = ex4.z / dn4.z;
    av[3] = ex4.w / dn4.w;
    if (lane == 0) {
        float4 a4 = make_float4(av[0], av[1], av[2], av[3]);
        ((float4*)a_out)[edge] = a4;
    }
    const int i0 = lane * 8;           /* slice start within [0,256) */
    const float a = av[lane >> 3];     /* head for this slice */
    const float4* fp = (const float4*)(feat_src + (long long)s * HD + i0);
    float4 f0 = fp[0];
    float4 f1 = fp[1];
    float* rp = rst + (long long)d * HD + i0;
    atomicAdd(rp + 0, a * f0.x);
    atomicAdd(rp + 1, a * f0.y);
    atomicAdd(rp + 2, a * f0.z);
    atomicAdd(rp + 3, a * f0.w);
    atomicAdd(rp + 4, a * f1.x);
    atomicAdd(rp + 5, a * f1.y);
    atomicAdd(rp + 6, a * f1.z);
    atomicAdd(rp + 7, a * f1.w);
}

extern "C" void kernel_launch(void* const* d_in, const int* in_sizes, int n_in,
                              void* d_out, int out_size, void* d_ws, size_t ws_size,
                              hipStream_t stream) {
    const float* feat       = (const float*)d_in[0];
    const int*   e_feat     = (const int*)  d_in[1];
    const int*   src        = (const int*)  d_in[2];
    const int*   dst        = (const int*)  d_in[3];
    const float* fc_w       = (const float*)d_in[4];
    const float* fc_e_w     = (const float*)d_in[5];
    const float* edge_emb_w = (const float*)d_in[6];
    const float* attn_l     = (const float*)d_in[7];
    const float* attn_r     = (const float*)d_in[8];
    const float* attn_e     = (const float*)d_in[9];

    float* ws       = (float*)d_ws;
    float* feat_src = ws + WS_FEATSRC;
    float* el       = ws + WS_EL;
    float* er       = ws + WS_ER;
    float* mmax     = ws + WS_M;
    float* den      = ws + WS_DEN;
    float* tbl      = ws + WS_TBL;

    float* rst   = (float*)d_out;                       /* [N, H, D] */
    float* a_out = (float*)d_out + (long long)N_NODES * HD; /* [E, H] */

    /* 1) collapse edge-type path into 8x4 table */
    gat55_table_kernel<<<1, 32, 0, stream>>>(fc_e_w, edge_emb_w, attn_e, tbl);

    /* 2) init m/denom/rst */
    {
        long long total = (long long)N_NODES * HD;
        int blocks = (int)((total + 255) / 256);
        gat55_init_kernel<<<blocks, 256, 0, stream>>>(mmax, den, rst);
    }

    /* 3) node projection via WMMA bf16 */
    {
        int waves = 3125 * 4;
        int blocks = (waves + 7) / 8;
        gat55_gemm_wmma<<<blocks, 256, 0, stream>>>(feat, fc_w, feat_src);
    }

    /* 4) el / er */
    {
        int total = N_NODES * HEADS;
        gat55_elr_kernel<<<(total + 255) / 256, 256, 0, stream>>>(
            feat_src, attn_l, attn_r, el, er);
    }

    /* 5) edge softmax: max pass */
    gat55_edge_max<<<(N_EDGES + 255) / 256, 256, 0, stream>>>(
        e_feat, src, dst, el, er, tbl, mmax);

    /* 6) edge softmax: exp + denom pass */
    gat55_edge_exp<<<(N_EDGES + 255) / 256, 256, 0, stream>>>(
        e_feat, src, dst, el, er, tbl, mmax, den, a_out);

    /* 7) normalize + weighted aggregation (wave per edge) */
    {
        int blocks = (N_EDGES + 7) / 8;
        gat55_edge_agg<<<blocks, 256, 0, stream>>>(
            src, dst, feat_src, den, a_out, rst);
    }
    (void)in_sizes; (void)n_in; (void)out_size; (void)ws_size;
}
